// VQVAE_60644938219912
// MI455X (gfx1250) — compile-verified, matrix operand order checked by reading, not run on hardware
//
#include <hip/hip_runtime.h>
#include <hip/hip_bf16.h>
#include <math.h>

// ---------------------------------------------------------------------------
// VQ-VAE forward for MI455X (gfx1250, wave32, WMMA + async-to-LDS staging).
// 4x4 convs / deconvs = implicit GEMM on v_wmma_f32_16x16x32_f16:
//   M = Cout (padded to 16), N = 16 output columns per wave, K = Cin*16.
// Input tiles staged into LDS with GLOBAL_LOAD_ASYNC_TO_LDS_B32 (ASYNCcnt),
// f32 in LDS, converted to f16 while building B fragments.
// ---------------------------------------------------------------------------

typedef __attribute__((ext_vector_type(16))) _Float16 v16h;
typedef __attribute__((ext_vector_type(8)))  _Float16 v8h;
typedef __attribute__((ext_vector_type(8)))  float    v8f;

// ---------------------------------------------------------------------------
// Generic WMMA conv kernel. Block: 128 threads = 4 waves, 64 output columns
// of one (n, oy) output row. Each wave: 16(co) x 16(ox) tile.
//   DECONV=false: out[n,co,oy,ox] = sum_{ci,ky,kx} w[co,ci,ky,kx]*in[n,ci,2oy-1+ky,2ox-1+kx]
//   DECONV=true : out[n,co,oy,ox] = sum w[ci,co,3-ky,3-kx]*xdil[n,ci,oy-2+ky,ox-2+kx]
// EPI: 0 = write raw (+bias), 1 = sigmoid + write + accumulate MSE vs xref.
// ---------------------------------------------------------------------------
template <int CIN, int COUT, bool DECONV, int EPI>
__global__ __launch_bounds__(128) void conv_wmma(
    const float* __restrict__ in, int Hin, int Win,
    const float* __restrict__ w, const float* __restrict__ bias,
    float* __restrict__ out, int Hout, int Wout,
    const float* __restrict__ xref, float* __restrict__ lossAcc)
{
    constexpr int K  = CIN * 16;            // Cin * 4 * 4
    constexpr int NK = (K + 31) / 32;       // WMMA K-steps
    constexpr int KP = NK * 32;             // padded K
    constexpr int TW = DECONV ? 68 : 130;   // LDS tile width (cols) for 64 out cols

    __shared__ __align__(16) _Float16 wl[16 * KP];  // A-matrix weights (f16)
    __shared__ float  tile[CIN][4][TW];             // [ci][ky][col] input tile (f32)
    __shared__ float  rbuf[128];                    // loss reduction

    const int n   = blockIdx.z;
    const int oy  = blockIdx.y;
    const int oxb = blockIdx.x * 64;
    const int tid = threadIdx.x;

    __builtin_prefetch(in, 0, 0);   // global_prefetch_b8

    // ---- weights -> LDS (f16), zero-padded to [16][KP] -------------------
    for (int i = tid; i < 16 * KP; i += 128) {
        const int co = i / KP, k = i % KP;
        float v = 0.f;
        if (co < COUT && k < K) {
            const int ci = k >> 4, rem = k & 15, ky = rem >> 2, kx = rem & 3;
            if (!DECONV) v = w[(co * CIN + ci) * 16 + ky * 4 + kx];
            else         v = w[(ci * COUT + co) * 16 + (3 - ky) * 4 + (3 - kx)];
        }
        wl[i] = (_Float16)v;
    }

    // ---- input tile: zero-fill, then async copy valid lanes --------------
    constexpr int TEL = CIN * 4 * TW;
    float* tf = &tile[0][0][0];
    for (int i = tid; i < TEL; i += 128) tf[i] = 0.f;
    __syncthreads();   // zeros visible before async engine writes land

    for (int i = tid; i < TEL; i += 128) {
        const int col = i % TW;
        const int r   = (i / TW) & 3;
        const int ci  = i / (4 * TW);
        bool valid = false;
        size_t gidx = 0;
        if (!DECONV) {
            const int iy = 2 * oy - 1 + r;
            const int ix = 2 * oxb - 1 + col;
            valid = (iy >= 0 && iy < Hin && ix >= 0 && ix < Win);
            if (valid) gidx = (((size_t)n * CIN + ci) * Hin + iy) * Win + ix;
        } else {
            const int dy = oy - 2 + r, dx = oxb - 2 + col;
            if (dy >= 0 && dx >= 0 && !(dy & 1) && !(dx & 1)) {
                const int iy = dy >> 1, ix = dx >> 1;
                valid = (iy < Hin && ix < Win);
                if (valid) gidx = (((size_t)n * CIN + ci) * Hin + iy) * Win + ix;
            }
        }
        if (valid) {
            // per-lane async DMA: LDS[vdst] = MEM[vaddr], tracked by ASYNCcnt
            unsigned lds = (unsigned)(unsigned long long)(&tf[i]);
            unsigned long long ga = (unsigned long long)(const void*)&in[gidx];
            asm volatile("global_load_async_to_lds_b32 %0, %1, off"
                         :: "v"(lds), "v"(ga) : "memory");
        }
    }
    asm volatile("s_wait_asynccnt 0x0" ::: "memory");
    __syncthreads();

    const int wave = tid >> 5;
    const int lane = tid & 31;
    const int half = lane >> 4;     // which 16-lane half
    const int j    = lane & 15;     // A row (m=co) / B col (output x)
    const int jj   = wave * 16 + j; // block-local output column (0..63)

    const v8h* wv = (const v8h*)wl;         // 16-byte aligned chunks
    const int rowbase = j * (KP / 8);

    v8f acc = {};
#pragma unroll
    for (int kk = 0; kk < NK; ++kk) {
        // A fragment: two aligned 16B LDS reads (ds_load_b128)
        const v8h a0 = wv[rowbase + kk * 4 + half];
        const v8h a1 = wv[rowbase + kk * 4 + 2 + half];
        v16h a, b;
#pragma unroll
        for (int e = 0; e < 8; ++e) { a[e] = a0[e]; a[8 + e] = a1[e]; }

        // B fragment: 4 groups of 4 contiguous f32 tile elements -> f16
#pragma unroll
        for (int g = 0; g < 4; ++g) {
            const int ci = kk * 2 + (g >> 1);
            const int ky = 2 * half + (g & 1);
#pragma unroll
            for (int t = 0; t < 4; ++t) {
                float v = 0.f;
                if (ci < CIN) {
                    const int col = DECONV ? (jj + t) : (2 * jj + t);
                    v = tile[ci < CIN ? ci : 0][ky][col];
                }
                b[g * 4 + t] = (_Float16)v;
            }
        }
        acc = __builtin_amdgcn_wmma_f32_16x16x32_f16(
            false, a, false, b, (short)0, acc, false, false);
    }

    // ---- epilogue --------------------------------------------------------
    const int ocol = oxb + jj;
    if constexpr (EPI == 0) {
#pragma unroll
        for (int v = 0; v < 8; ++v) {
            const int co = half * 8 + v;
            if (co < COUT && ocol < Wout) {
                const size_t idx = (((size_t)n * COUT + co) * Hout + oy) * Wout + ocol;
                out[idx] = acc[v] + bias[co];
            }
        }
    } else {
        float lsum = 0.f;
#pragma unroll
        for (int v = 0; v < 8; ++v) {
            const int co = half * 8 + v;
            if (co < COUT && ocol < Wout) {
                const size_t idx = (((size_t)n * COUT + co) * Hout + oy) * Wout + ocol;
                const float z = acc[v] + bias[co];
                const float y = 1.f / (1.f + __expf(-z));
                out[idx] = y;
                const float d = xref[idx] - y;
                lsum += d * d;
            }
        }
        rbuf[tid] = lsum;
        __syncthreads();
        for (int s = 64; s > 0; s >>= 1) {
            if (tid < s) rbuf[tid] += rbuf[tid + s];
            __syncthreads();
        }
        if (tid == 0) atomicAdd(lossAcc, rbuf[0]);
    }
}

// ---------------------------------------------------------------------------
// Per-channel sum / sumsq reduction (for BatchNorm), NCHW layout.
// st[0..C) = sum, st[C..2C) = sumsq.
// ---------------------------------------------------------------------------
template <int C>
__global__ __launch_bounds__(256) void stats_kernel(
    const float* __restrict__ x, long total, int HW, float* __restrict__ st)
{
    __shared__ float ssum[C], ssq[C];
    for (int i = threadIdx.x; i < C; i += blockDim.x) { ssum[i] = 0.f; ssq[i] = 0.f; }
    __syncthreads();
    for (long i = blockIdx.x * (long)blockDim.x + threadIdx.x; i < total;
         i += (long)gridDim.x * blockDim.x) {
        const float v = x[i];
        const int c = (int)((i / HW) % C);
        atomicAdd(&ssum[c], v);
        atomicAdd(&ssq[c], v * v);
    }
    __syncthreads();
    for (int i = threadIdx.x; i < C; i += blockDim.x) {
        atomicAdd(&st[i], ssum[i]);
        atomicAdd(&st[C + i], ssq[i]);
    }
}

// ---------------------------------------------------------------------------
// In-place BatchNorm (training stats) + ReLU.
// ---------------------------------------------------------------------------
__global__ __launch_bounds__(256) void bnrelu_kernel(
    float* __restrict__ x, long total, int HW, int C,
    const float* __restrict__ st, const float* __restrict__ g,
    const float* __restrict__ be, float invcnt)
{
    for (long i = blockIdx.x * (long)blockDim.x + threadIdx.x; i < total;
         i += (long)gridDim.x * blockDim.x) {
        const int c = (int)((i / HW) % C);
        const float m   = st[c] * invcnt;
        const float var = st[C + c] * invcnt - m * m;
        const float v = (x[i] - m) * rsqrtf(var + 1e-5f) * g[c] + be[c];
        x[i] = v > 0.f ? v : 0.f;
    }
}

// ---------------------------------------------------------------------------
// 1x1 conv (tiny channel counts).
// ---------------------------------------------------------------------------
template <int CIN, int COUT>
__global__ __launch_bounds__(256) void conv1x1_kernel(
    const float* __restrict__ in, const float* __restrict__ w,
    const float* __restrict__ b, float* __restrict__ out, int HW, int N)
{
    const long total = (long)N * HW;
    for (long i = blockIdx.x * (long)blockDim.x + threadIdx.x; i < total;
         i += (long)gridDim.x * blockDim.x) {
        const int n = (int)(i / HW), s = (int)(i % HW);
        float iv[CIN];
#pragma unroll
        for (int ci = 0; ci < CIN; ++ci)
            iv[ci] = in[((long)(n * CIN + ci)) * HW + s];
#pragma unroll
        for (int co = 0; co < COUT; ++co) {
            float a = b[co];
#pragma unroll
            for (int ci = 0; ci < CIN; ++ci) a += w[co * CIN + ci] * iv[ci];
            out[((long)(n * COUT + co)) * HW + s] = a;
        }
    }
}

// ---------------------------------------------------------------------------
// Vector quantization: argmin over 256 2-d codes; writes quantized field and
// accumulates sum((quant - qf)^2) into acc[0].
// ---------------------------------------------------------------------------
__global__ __launch_bounds__(256) void vq_kernel(
    const float* __restrict__ q, const float* __restrict__ cb,
    float* __restrict__ quant, float* __restrict__ acc, int HW, int N)
{
    __shared__ float c0s[256], c1s[256];
    __shared__ float rbuf[256];
    for (int i = threadIdx.x; i < 256; i += blockDim.x) {
        c0s[i] = cb[2 * i];
        c1s[i] = cb[2 * i + 1];
    }
    __syncthreads();
    float lsum = 0.f;
    const long total = (long)N * HW;
    for (long i = blockIdx.x * (long)blockDim.x + threadIdx.x; i < total;
         i += (long)gridDim.x * blockDim.x) {
        const int n = (int)(i / HW), s = (int)(i % HW);
        const float q0 = q[((long)(n * 2 + 0)) * HW + s];
        const float q1 = q[((long)(n * 2 + 1)) * HW + s];
        float bd = 3.4e38f;
        int bi = 0;
#pragma unroll 4
        for (int c = 0; c < 256; ++c) {
            const float d0 = q0 - c0s[c], d1 = q1 - c1s[c];
            const float d = d0 * d0 + d1 * d1;
            if (d < bd) { bd = d; bi = c; }
        }
        quant[((long)(n * 2 + 0)) * HW + s] = c0s[bi];
        quant[((long)(n * 2 + 1)) * HW + s] = c1s[bi];
        lsum += bd;
    }
    rbuf[threadIdx.x] = lsum;
    __syncthreads();
    for (int s = 128; s > 0; s >>= 1) {
        if (threadIdx.x < (unsigned)s) rbuf[threadIdx.x] += rbuf[threadIdx.x + s];
        __syncthreads();
    }
    if (threadIdx.x == 0) atomicAdd(acc, rbuf[0]);
}

__global__ void zero_kernel(float* __restrict__ p, int n)
{
    const int i = blockIdx.x * blockDim.x + threadIdx.x;
    if (i < n) p[i] = 0.f;
}

// losses[0] = sum (quant-qf)^2  over 262144 elems
// losses[1] = sum (x-out)^2     over 8388608 elems
// total = recon + codebook + 0.2*commitment = recon_mean + 1.2*vq_mean
__global__ void finalize_kernel(float* __restrict__ out, const float* __restrict__ losses)
{
    if (threadIdx.x == 0 && blockIdx.x == 0)
        out[8388608] = losses[1] * (1.f / 8388608.f) + 1.2f * losses[0] * (1.f / 262144.f);
}

// ---------------------------------------------------------------------------
extern "C" void kernel_launch(void* const* d_in, const int* in_sizes, int n_in,
                              void* d_out, int out_size, void* d_ws, size_t ws_size,
                              hipStream_t stream)
{
    const float* x    = (const float*)d_in[0];
    const float* w1   = (const float*)d_in[1];
    const float* b1   = (const float*)d_in[2];
    const float* g1   = (const float*)d_in[3];
    const float* be1  = (const float*)d_in[4];
    const float* w2   = (const float*)d_in[5];
    const float* b2   = (const float*)d_in[6];
    const float* g2   = (const float*)d_in[7];
    const float* be2  = (const float*)d_in[8];
    const float* w3   = (const float*)d_in[9];
    const float* b3   = (const float*)d_in[10];
    const float* g3   = (const float*)d_in[11];
    const float* be3  = (const float*)d_in[12];
    const float* wpre = (const float*)d_in[13];
    const float* bpre = (const float*)d_in[14];
    const float* cb   = (const float*)d_in[15];
    const float* wpost= (const float*)d_in[16];
    const float* bpost= (const float*)d_in[17];
    const float* wd1  = (const float*)d_in[18];
    const float* bd1  = (const float*)d_in[19];
    const float* g4   = (const float*)d_in[20];
    const float* be4  = (const float*)d_in[21];
    const float* wd2  = (const float*)d_in[22];
    const float* bd2  = (const float*)d_in[23];
    const float* g5   = (const float*)d_in[24];
    const float* be5  = (const float*)d_in[25];
    const float* wd3  = (const float*)d_in[26];
    const float* bd3  = (const float*)d_in[27];

    float* outp = (float*)d_out;

    // workspace layout (floats); big buffers ping-pong encoder/decoder
    float* A   = (float*)d_ws;            // 32*16*256*256 = 33554432 (h1 / d2)
    float* B   = A   + 33554432;          // 32*16*128*128 =  8388608 (h2 / d1)
    float* C3  = B   + 8388608;           // 32*4*64*64    =   524288 (h3)
    float* Q   = C3  + 524288;            // 32*2*64*64    =   262144
    float* QT  = Q   + 262144;            //                  262144 (quant)
    float* D0  = QT  + 262144;            // 32*4*64*64    =   524288
    float* ST  = D0  + 524288;            // stats: 5 slots * 32 + losses

    const dim3 blk(128);

    // zero BN stats + loss accumulators (ws is poisoned once, never re-poisoned)
    zero_kernel<<<1, 256, 0, stream>>>(ST, 192);

    // ---- encoder ---------------------------------------------------------
    conv_wmma<1, 16, false, 0><<<dim3(4, 256, 32), blk, 0, stream>>>(
        x, 512, 512, w1, b1, A, 256, 256, nullptr, nullptr);
    stats_kernel<16><<<1024, 256, 0, stream>>>(A, 33554432L, 65536, ST + 0);
    bnrelu_kernel<<<1024, 256, 0, stream>>>(A, 33554432L, 65536, 16, ST + 0, g1, be1, 1.f / 2097152.f);

    conv_wmma<16, 16, false, 0><<<dim3(2, 128, 32), blk, 0, stream>>>(
        A, 256, 256, w2, b2, B, 128, 128, nullptr, nullptr);
    stats_kernel<16><<<1024, 256, 0, stream>>>(B, 8388608L, 16384, ST + 32);
    bnrelu_kernel<<<1024, 256, 0, stream>>>(B, 8388608L, 16384, 16, ST + 32, g2, be2, 1.f / 524288.f);

    conv_wmma<16, 4, false, 0><<<dim3(1, 64, 32), blk, 0, stream>>>(
        B, 128, 128, w3, b3, C3, 64, 64, nullptr, nullptr);
    stats_kernel<4><<<512, 256, 0, stream>>>(C3, 524288L, 4096, ST + 64);
    bnrelu_kernel<<<512, 256, 0, stream>>>(C3, 524288L, 4096, 4, ST + 64, g3, be3, 1.f / 131072.f);

    // ---- vector quantization --------------------------------------------
    conv1x1_kernel<4, 2><<<512, 256, 0, stream>>>(C3, wpre, bpre, Q, 4096, 32);
    vq_kernel<<<512, 256, 0, stream>>>(Q, cb, QT, ST + 160, 4096, 32);
    conv1x1_kernel<2, 4><<<512, 256, 0, stream>>>(QT, wpost, bpost, D0, 4096, 32);

    // ---- decoder ---------------------------------------------------------
    conv_wmma<4, 16, true, 0><<<dim3(2, 128, 32), blk, 0, stream>>>(
        D0, 64, 64, wd1, bd1, B, 128, 128, nullptr, nullptr);
    stats_kernel<16><<<1024, 256, 0, stream>>>(B, 8388608L, 16384, ST + 96);
    bnrelu_kernel<<<1024, 256, 0, stream>>>(B, 8388608L, 16384, 16, ST + 96, g4, be4, 1.f / 524288.f);

    conv_wmma<16, 16, true, 0><<<dim3(4, 256, 32), blk, 0, stream>>>(
        B, 128, 128, wd2, bd2, A, 256, 256, nullptr, nullptr);
    stats_kernel<16><<<1024, 256, 0, stream>>>(A, 33554432L, 65536, ST + 128);
    bnrelu_kernel<<<1024, 256, 0, stream>>>(A, 33554432L, 65536, 16, ST + 128, g5, be5, 1.f / 2097152.f);

    // final deconv fused with sigmoid + recon-MSE accumulation
    conv_wmma<16, 1, true, 1><<<dim3(8, 512, 32), blk, 0, stream>>>(
        A, 256, 256, wd3, bd3, outp, 512, 512, x, ST + 161);

    finalize_kernel<<<1, 1, 0, stream>>>(outp, ST + 160);
}
